// Decoder_86114094284949
// MI455X (gfx1250) — matrix operand order repaired and checked by reading, not compiled, not run on hardware
//
#include <hip/hip_runtime.h>
#include <math.h>

typedef __bf16 bf16_t;
typedef __attribute__((ext_vector_type(16))) bf16_t v16bf;
typedef __attribute__((ext_vector_type(8)))  bf16_t v8bf;
typedef __attribute__((ext_vector_type(8)))  float  v8f;

#define L_ORI 6
#define BATCH 2

// ---------------------------------------------------------------------------
// prep: xin = concat(unpool(prev), enc)  ->  Tt[b][n][c] (bf16, T0 section)
// ---------------------------------------------------------------------------
__global__ void prep_concat_kernel(const float* __restrict__ prev,
                                   const float* __restrict__ enc,
                                   bf16_t* __restrict__ Tt,
                                   int N, int cin, int cprev, int Kpad, int Nprev) {
  int idx = blockIdx.x * blockDim.x + threadIdx.x;
  int total = BATCH * N * cin;
  if (idx >= total) return;
  int c = idx % cin;
  int n = (idx / cin) % N;
  int b = idx / (cin * N);
  int Vp = Nprev / L_ORI;
  int r  = n / (2 * Vp);
  int v2 = n - r * (2 * Vp);
  int np = r * Vp + (v2 >> 1);
  int cenc = cin - cprev;
  float v;
  if (c < cprev) v = prev[((size_t)b * cprev + c) * (size_t)Nprev + np];
  else           v = enc[((size_t)b * cenc + (c - cprev)) * (size_t)N + n];
  Tt[((size_t)b * N + n) * (size_t)Kpad + c] = (bf16_t)v;
}

// prep: copy h (b,C,N channel-major f32) -> Tt T0 section (node-major bf16)
__global__ void prep_copy_kernel(const float* __restrict__ h,
                                 bf16_t* __restrict__ Tt,
                                 int N, int C, int Kpad) {
  int idx = blockIdx.x * blockDim.x + threadIdx.x;
  int total = BATCH * N * C;
  if (idx >= total) return;
  int c = idx % C;
  int n = (idx / C) % N;
  int b = idx / (C * N);
  float v = h[((size_t)b * C + c) * (size_t)N + n];
  Tt[((size_t)b * N + n) * (size_t)Kpad + c] = (bf16_t)v;
}

// zero-fill K padding tail [kstart, Kpad) of every row
__global__ void pad_zero_kernel(bf16_t* __restrict__ Tt, int rows, int kstart, int Kpad) {
  int pad = Kpad - kstart;
  int idx = blockIdx.x * blockDim.x + threadIdx.x;
  int total = rows * pad;
  if (idx >= total) return;
  int row = idx / pad;
  int kk  = kstart + idx % pad;
  Tt[(size_t)row * Kpad + kk] = (bf16_t)0.0f;
}

// ---------------------------------------------------------------------------
// Chebyshev recursion on the circulant graph (gather form, exploits
// dst[o*N+i] == i layout of the edge list).
// mode 0: T_out = L*T_in            (T1 = L*T0)
// mode 1: T_out = 2*L*T_in - T0    (T2)
// ---------------------------------------------------------------------------
__global__ void lap_kernel(bf16_t* __restrict__ Tt,
                           const int* __restrict__ src,
                           const float* __restrict__ cval,
                           int N, int C, int Kpad, int inOff, int outOff, int mode) {
  int idx = blockIdx.x * blockDim.x + threadIdx.x;
  int total = BATCH * N * C;
  if (idx >= total) return;
  int c = idx % C;
  int n = (idx / C) % N;
  int b = idx / (C * N);
  size_t bbase = (size_t)b * N * (size_t)Kpad;
  float acc = 0.0f;
#pragma unroll
  for (int o = 0; o < 8; ++o) {
    int e = o * N + n;
    int s = src[e];
    acc += cval[e] * (float)Tt[bbase + (size_t)s * Kpad + inOff + c];
  }
  if (mode) acc = 2.0f * acc - (float)Tt[bbase + (size_t)n * Kpad + c];
  Tt[bbase + (size_t)n * Kpad + outOff + c] = (bf16_t)acc;
}

// weights (K*cin, cout) f32 -> Wt (cout, Kpad) bf16, zero-padded
__global__ void wprep_kernel(const float* __restrict__ w, bf16_t* __restrict__ Wt,
                             int Kcin, int cout, int Kpad) {
  int idx = blockIdx.x * blockDim.x + threadIdx.x;
  int total = cout * Kpad;
  if (idx >= total) return;
  int kk = idx % Kpad;
  int o  = idx / Kpad;
  float v = (kk < Kcin) ? w[(size_t)kk * cout + o] : 0.0f;
  Wt[(size_t)o * Kpad + kk] = (bf16_t)v;
}

// ---------------------------------------------------------------------------
// WMMA GEMM helpers
// ---------------------------------------------------------------------------
__device__ __forceinline__ v16bf load_a_frag(const bf16_t* Arow, int kb, int half) {
  union { v16bf v; v8bf h[2]; } au;
  au.h[0] = *(const v8bf*)(Arow + kb + half * 16);
  au.h[1] = *(const v8bf*)(Arow + kb + half * 16 + 8);
  return au.v;
}

// ---------------------------------------------------------------------------
// WMMA GEMM (fat tile): C[b,m,n] = sum_k A[m,k]*Bt[b,n,k] (+bias[m])
// 4 waves/block; each wave owns a 32(M) x 64(N) tile = 8 f32 accumulators.
// Per k-step: 12 b128 loads -> 8 v_wmma_f32_16x16x32_bf16 (1.5 loads/WMMA).
// Requires M % 32 == 0.
// ---------------------------------------------------------------------------
__global__ __launch_bounds__(128)
void wmma_gemm32_kernel(const bf16_t* __restrict__ A,
                        const bf16_t* __restrict__ Bt,
                        const float* __restrict__ bias,
                        float* __restrict__ C,
                        int M, int Nn, int Kd, int lda, int ldb) {
  int lane  = threadIdx.x & 31;
  int wave  = threadIdx.x >> 5;
  int b     = blockIdx.z;
  int mBase = blockIdx.y * 32;
  int nBase = blockIdx.x * 256 + wave * 64;
  int half  = lane >> 4;
  int l16   = lane & 15;

  const bf16_t* Bb    = Bt + (size_t)b * Nn * (size_t)ldb;
  const bf16_t* Arow0 = A + (size_t)(mBase + l16) * lda;
  const bf16_t* Arow1 = A + (size_t)(mBase + 16 + l16) * lda;
  const bf16_t* Brow0 = Bb + (size_t)(nBase +  0 + l16) * ldb;
  const bf16_t* Brow1 = Bb + (size_t)(nBase + 16 + l16) * ldb;
  const bf16_t* Brow2 = Bb + (size_t)(nBase + 32 + l16) * ldb;
  const bf16_t* Brow3 = Bb + (size_t)(nBase + 48 + l16) * ldb;

  v8f a00 = {}, a01 = {}, a02 = {}, a03 = {};
  v8f a10 = {}, a11 = {}, a12 = {}, a13 = {};

  for (int kb = 0; kb < Kd; kb += 32) {
    v16bf av0 = load_a_frag(Arow0, kb, half);
    v16bf av1 = load_a_frag(Arow1, kb, half);
    int ko = kb + half * 16;
    v16bf b0 = *(const v16bf*)(Brow0 + ko);
    v16bf b1 = *(const v16bf*)(Brow1 + ko);
    v16bf b2 = *(const v16bf*)(Brow2 + ko);
    v16bf b3 = *(const v16bf*)(Brow3 + ko);
    a00 = __builtin_amdgcn_wmma_f32_16x16x32_bf16(false, av0, false, b0, (short)0, a00, false, false);
    a01 = __builtin_amdgcn_wmma_f32_16x16x32_bf16(false, av0, false, b1, (short)0, a01, false, false);
    a02 = __builtin_amdgcn_wmma_f32_16x16x32_bf16(false, av0, false, b2, (short)0, a02, false, false);
    a03 = __builtin_amdgcn_wmma_f32_16x16x32_bf16(false, av0, false, b3, (short)0, a03, false, false);
    a10 = __builtin_amdgcn_wmma_f32_16x16x32_bf16(false, av1, false, b0, (short)0, a10, false, false);
    a11 = __builtin_amdgcn_wmma_f32_16x16x32_bf16(false, av1, false, b1, (short)0, a11, false, false);
    a12 = __builtin_amdgcn_wmma_f32_16x16x32_bf16(false, av1, false, b2, (short)0, a12, false, false);
    a13 = __builtin_amdgcn_wmma_f32_16x16x32_bf16(false, av1, false, b3, (short)0, a13, false, false);
  }

  float* Cb = C + (size_t)b * M * (size_t)Nn;
#pragma unroll
  for (int j = 0; j < 8; ++j) {
    int m0 = mBase + half * 8 + j;
    int m1 = m0 + 16;
    float bv0 = bias ? bias[m0] : 0.0f;
    float bv1 = bias ? bias[m1] : 0.0f;
    size_t rb0 = (size_t)m0 * Nn + nBase + l16;
    size_t rb1 = (size_t)m1 * Nn + nBase + l16;
    Cb[rb0 +  0] = a00[j] + bv0;
    Cb[rb0 + 16] = a01[j] + bv0;
    Cb[rb0 + 32] = a02[j] + bv0;
    Cb[rb0 + 48] = a03[j] + bv0;
    Cb[rb1 +  0] = a10[j] + bv1;
    Cb[rb1 + 16] = a11[j] + bv1;
    Cb[rb1 + 32] = a12[j] + bv1;
    Cb[rb1 + 48] = a13[j] + bv1;
  }
}

// ---------------------------------------------------------------------------
// WMMA GEMM (slim tile, M==16 fallback): 16(M) x 64(N) per wave.
// ---------------------------------------------------------------------------
__global__ __launch_bounds__(128)
void wmma_gemm_kernel(const bf16_t* __restrict__ A,
                      const bf16_t* __restrict__ Bt,
                      const float* __restrict__ bias,
                      float* __restrict__ C,
                      int M, int Nn, int Kd, int lda, int ldb) {
  int lane  = threadIdx.x & 31;
  int wave  = threadIdx.x >> 5;
  int b     = blockIdx.z;
  int mBase = blockIdx.y * 16;
  int nBase = blockIdx.x * 256 + wave * 64;
  int half  = lane >> 4;
  int l16   = lane & 15;

  const bf16_t* Bb    = Bt + (size_t)b * Nn * (size_t)ldb;
  const bf16_t* Arow  = A + (size_t)(mBase + l16) * lda;
  const bf16_t* Brow0 = Bb + (size_t)(nBase +  0 + l16) * ldb;
  const bf16_t* Brow1 = Bb + (size_t)(nBase + 16 + l16) * ldb;
  const bf16_t* Brow2 = Bb + (size_t)(nBase + 32 + l16) * ldb;
  const bf16_t* Brow3 = Bb + (size_t)(nBase + 48 + l16) * ldb;

  v8f acc0 = {}, acc1 = {}, acc2 = {}, acc3 = {};

  for (int kb = 0; kb < Kd; kb += 32) {
    v16bf av = load_a_frag(Arow, kb, half);
    int ko = kb + half * 16;
    v16bf b0 = *(const v16bf*)(Brow0 + ko);
    v16bf b1 = *(const v16bf*)(Brow1 + ko);
    v16bf b2 = *(const v16bf*)(Brow2 + ko);
    v16bf b3 = *(const v16bf*)(Brow3 + ko);
    acc0 = __builtin_amdgcn_wmma_f32_16x16x32_bf16(false, av, false, b0, (short)0, acc0, false, false);
    acc1 = __builtin_amdgcn_wmma_f32_16x16x32_bf16(false, av, false, b1, (short)0, acc1, false, false);
    acc2 = __builtin_amdgcn_wmma_f32_16x16x32_bf16(false, av, false, b2, (short)0, acc2, false, false);
    acc3 = __builtin_amdgcn_wmma_f32_16x16x32_bf16(false, av, false, b3, (short)0, acc3, false, false);
  }

  float* Cb = C + (size_t)b * M * (size_t)Nn;
#pragma unroll
  for (int j = 0; j < 8; ++j) {
    int m = mBase + half * 8 + j;
    float bv = bias ? bias[m] : 0.0f;
    size_t rb = (size_t)m * Nn + nBase + l16;
    Cb[rb +  0] = acc0[j] + bv;
    Cb[rb + 16] = acc1[j] + bv;
    Cb[rb + 32] = acc2[j] + bv;
    Cb[rb + 48] = acc3[j] + bv;
  }
}

// dispatch helper (host side)
static inline void launch_gemm(const bf16_t* A, const bf16_t* Bt, const float* bias,
                               float* C, int M, int Nn, int Kd, int lda, int ldb,
                               hipStream_t stream) {
  if ((M & 31) == 0) {
    dim3 g(Nn / 256, M / 32, BATCH);
    wmma_gemm32_kernel<<<g, 128, 0, stream>>>(A, Bt, bias, C, M, Nn, Kd, lda, ldb);
  } else {
    dim3 g(Nn / 256, M / 16, BATCH);
    wmma_gemm_kernel<<<g, 128, 0, stream>>>(A, Bt, bias, C, M, Nn, Kd, lda, ldb);
  }
}

// ---------------------------------------------------------------------------
// BatchNorm stats: one block per channel, reduce over (B, N)
// ---------------------------------------------------------------------------
__global__ void stats_kernel(const float* __restrict__ X, float* __restrict__ stats,
                             int M, int Nn) {
  int o = blockIdx.x;
  __shared__ float ss[256];
  __shared__ float sq[256];
  float s = 0.0f, q = 0.0f;
  int cnt = BATCH * Nn;
  for (int i = threadIdx.x; i < cnt; i += blockDim.x) {
    int b = i / Nn;
    int n = i - b * Nn;
    float v = X[((size_t)b * M + o) * (size_t)Nn + n];
    s += v; q += v * v;
  }
  ss[threadIdx.x] = s; sq[threadIdx.x] = q;
  __syncthreads();
  for (int st = 128; st > 0; st >>= 1) {
    if (threadIdx.x < st) { ss[threadIdx.x] += ss[threadIdx.x + st]; sq[threadIdx.x] += sq[threadIdx.x + st]; }
    __syncthreads();
  }
  if (threadIdx.x == 0) {
    float c = (float)cnt;
    float m = ss[0] / c;
    stats[2 * o]     = m;
    stats[2 * o + 1] = sq[0] / c - m * m;
  }
}

// BN + (optional shortcut add) + ReLU
__global__ void bn_kernel(const float* __restrict__ X, const float* __restrict__ stats,
                          const float* __restrict__ g, const float* __restrict__ be,
                          const float* __restrict__ sc, float* __restrict__ Y,
                          int M, int Nn, int mode) {
  int idx = blockIdx.x * blockDim.x + threadIdx.x;
  int total = BATCH * M * Nn;
  if (idx >= total) return;
  int o = (idx / Nn) % M;
  float m = stats[2 * o], v = stats[2 * o + 1];
  float y = g[o] * (X[idx] - m) * rsqrtf(v + 1e-5f) + be[o];
  if (mode) y += sc[idx];
  Y[idx] = fmaxf(y, 0.0f);
}

// pool5: (B,16,98304) -> max over L_ORI -> (B,16,16384)
__global__ void pool_kernel(const float* __restrict__ x, float* __restrict__ out) {
  int idx = blockIdx.x * blockDim.x + threadIdx.x;
  int total = BATCH * 16 * 16384;
  if (idx >= total) return;
  int v = idx % 16384;
  int c = (idx / 16384) % 16;
  int b = idx / (16 * 16384);
  float m = -3.4e38f;
#pragma unroll
  for (int r = 0; r < L_ORI; ++r)
    m = fmaxf(m, x[((size_t)(b * 16 + c)) * 98304 + r * 16384 + v]);
  out[idx] = m;
}

// 1x1 conv head (16 -> 10) + log_softmax over channel axis
__global__ void head_kernel(const float* __restrict__ pooled,
                            const float* __restrict__ wout,
                            float* __restrict__ out) {
  int idx = blockIdx.x * blockDim.x + threadIdx.x;
  int total = BATCH * 16384;
  if (idx >= total) return;
  int v = idx % 16384;
  int b = idx / 16384;
  float xv[16];
#pragma unroll
  for (int c = 0; c < 16; ++c)
    xv[c] = pooled[((size_t)b * 16 + c) * 16384 + v];
  float lg[10];
  float mx = -3.4e38f;
#pragma unroll
  for (int j = 0; j < 10; ++j) {
    float s = 0.0f;
#pragma unroll
    for (int c = 0; c < 16; ++c) s += xv[c] * wout[c * 10 + j];
    lg[j] = s;
    mx = fmaxf(mx, s);
  }
  float se = 0.0f;
#pragma unroll
  for (int j = 0; j < 10; ++j) se += __expf(lg[j] - mx);
  float lse = mx + __logf(se);
#pragma unroll
  for (int j = 0; j < 10; ++j)
    out[((size_t)b * 10 + j) * 16384 + v] = lg[j] - lse;
}

// ---------------------------------------------------------------------------
extern "C" void kernel_launch(void* const* d_in, const int* in_sizes, int n_in,
                              void* d_out, int out_size, void* d_ws, size_t ws_size,
                              hipStream_t stream) {
  (void)in_sizes; (void)n_in; (void)out_size; (void)ws_size;

  // input layout: 0..5 x_enc, 6..20 edges (src,dst,c)*5, 21..65 params (9*5), 66 w_out
  const float* enc[6];
  for (int i = 0; i < 6; ++i) enc[i] = (const float*)d_in[i];
  auto edge_src = [&](int l) { return (const int*)  d_in[6 + 3 * l]; };
  auto edge_c   = [&](int l) { return (const float*)d_in[8 + 3 * l]; };
  auto prm      = [&](int l, int j) { return (const float*)d_in[21 + 9 * l + j]; };
  const float* w_out = (const float*)d_in[66];

  static const int cins[5]   = {512, 512, 256, 128, 64};
  static const int couts[5]  = {256, 128, 64, 32, 16};
  static const int cprevs[5] = {256, 256, 128, 64, 32};
  static const int Ns[5]     = {6144, 12288, 24576, 49152, 98304};

  // workspace carve-up
  char* base = (char*)d_ws;
  size_t off = 0;
  auto alloc = [&](size_t bytes) {
    void* p = base + off;
    off = (off + bytes + 255) & ~(size_t)255;
    return p;
  };
  bf16_t* Tt   = (bf16_t*)alloc((size_t)2 * 12288 * 1536 * sizeof(bf16_t)); // 75.5 MB, max over blocks
  float*  Y    = (float*) alloc((size_t)2 * 1572864 * sizeof(float));       // 12.6 MB (cout*N const)
  float*  SC   = (float*) alloc((size_t)2 * 1572864 * sizeof(float));
  float*  H1   = (float*) alloc((size_t)2 * 1572864 * sizeof(float));
  float*  OA   = (float*) alloc((size_t)2 * 1572864 * sizeof(float));
  float*  OB   = (float*) alloc((size_t)2 * 1572864 * sizeof(float));
  bf16_t* Wt1  = (bf16_t*)alloc((size_t)256 * 1536 * sizeof(bf16_t));
  bf16_t* Wt2  = (bf16_t*)alloc((size_t)256 * 768 * sizeof(bf16_t));
  bf16_t* Wsc  = (bf16_t*)alloc((size_t)256 * 512 * sizeof(bf16_t));
  float*  stats= (float*) alloc(2 * 256 * sizeof(float));
  float*  pooled=(float*) alloc((size_t)2 * 16 * 16384 * sizeof(float));

  const int TB = 256;
  const float* prev = enc[0];
  float* outb[2] = {OA, OB};

  for (int l = 0; l < 5; ++l) {
    int N = Ns[l], cin = cins[l], cout = couts[l], cprev = cprevs[l];
    int Nprev = N / 2;
    int K1 = 3 * cin;  int Kp1 = (K1 + 31) & ~31;   // == K1 (all mult. of 32)
    int K2 = 3 * cout; int Kp2 = (K2 + 31) & ~31;   // pads 48 -> 64 on block 4
    const int* src = edge_src(l);
    const float* cv = edge_c(l);

    // weight preps
    {
      int t1 = cout * Kp1;
      wprep_kernel<<<(t1 + TB - 1) / TB, TB, 0, stream>>>(prm(l, 0), Wt1, 3 * cin, cout, Kp1);
      int t2 = cout * Kp2;
      wprep_kernel<<<(t2 + TB - 1) / TB, TB, 0, stream>>>(prm(l, 4), Wt2, 3 * cout, cout, Kp2);
      int t3 = cout * cin;
      wprep_kernel<<<(t3 + TB - 1) / TB, TB, 0, stream>>>(prm(l, 8), Wsc, cin, cout, cin);
    }

    // ---- cheb1 input: concat(unpool(prev), enc[l+1]) -> Tt T0; T1; T2
    int tot1 = BATCH * N * cin;
    prep_concat_kernel<<<(tot1 + TB - 1) / TB, TB, 0, stream>>>(prev, enc[l + 1], Tt, N, cin, cprev, Kp1, Nprev);
    lap_kernel<<<(tot1 + TB - 1) / TB, TB, 0, stream>>>(Tt, src, cv, N, cin, Kp1, 0, cin, 0);
    lap_kernel<<<(tot1 + TB - 1) / TB, TB, 0, stream>>>(Tt, src, cv, N, cin, Kp1, cin, 2 * cin, 1);

    // ---- GEMMs: Y = Wt1 x Tt + b1 ; SC = Wsc x T0
    launch_gemm(Wt1, Tt, prm(l, 1), Y, cout, N, Kp1, Kp1, Kp1, stream);
    launch_gemm(Wsc, Tt, nullptr,   SC, cout, N, cin, cin, Kp1, stream);

    // ---- BN1 + ReLU -> H1
    int toty = BATCH * cout * N;
    stats_kernel<<<cout, 256, 0, stream>>>(Y, stats, cout, N);
    bn_kernel<<<(toty + TB - 1) / TB, TB, 0, stream>>>(Y, stats, prm(l, 2), prm(l, 3), nullptr, H1, cout, N, 0);

    // ---- cheb2 on H1 (reuse Tt with stride Kp2)
    int tot2 = BATCH * N * cout;
    prep_copy_kernel<<<(tot2 + TB - 1) / TB, TB, 0, stream>>>(H1, Tt, N, cout, Kp2);
    if (Kp2 > 3 * cout) {
      int rows = BATCH * N, padtot = rows * (Kp2 - 3 * cout);
      pad_zero_kernel<<<(padtot + TB - 1) / TB, TB, 0, stream>>>(Tt, rows, 3 * cout, Kp2);
    }
    lap_kernel<<<(tot2 + TB - 1) / TB, TB, 0, stream>>>(Tt, src, cv, N, cout, Kp2, 0, cout, 0);
    lap_kernel<<<(tot2 + TB - 1) / TB, TB, 0, stream>>>(Tt, src, cv, N, cout, Kp2, cout, 2 * cout, 1);
    launch_gemm(Wt2, Tt, prm(l, 5), Y, cout, N, Kp2, Kp2, Kp2, stream);

    // ---- BN2 + shortcut + ReLU -> block output
    stats_kernel<<<cout, 256, 0, stream>>>(Y, stats, cout, N);
    bn_kernel<<<(toty + TB - 1) / TB, TB, 0, stream>>>(Y, stats, prm(l, 6), prm(l, 7), SC, outb[l & 1], cout, N, 1);
    prev = outb[l & 1];
  }

  // ---- pool over orientations + 1x1 head + log_softmax
  {
    int tp = BATCH * 16 * 16384;
    pool_kernel<<<(tp + TB - 1) / TB, TB, 0, stream>>>(prev, pooled);
    int th = BATCH * 16384;
    head_kernel<<<(th + TB - 1) / TB, TB, 0, stream>>>(pooled, w_out, (float*)d_out);
  }
}
